// AutoCorrelation_85134841742133
// MI455X (gfx1250) — matrix operand order repaired and checked by reading, not compiled, not run on hardware
//
#include <hip/hip_runtime.h>
#include <hip/hip_bf16.h>

// ---------------------------------------------------------------------------
// AutoCorrelation block for MI455X (gfx1250, wave32, WMMA).
//  x:[4,4096,1024] f32.  Pipeline:
//   1) convert x + 4 weight matrices to f16
//   2) Q/K/V projections: 128x128-tile WMMA f16 GEMM (f32 accum) + bias,
//      double-buffered LDS so global_load_b128 overlaps v_wmma
//   3) transpose Q,K to [B,C,S] for coalesced per-channel FFTs
//   4) per-channel 4096-pt Stockham FFT in LDS (Q+iK packed complex),
//      Hermitian split, P = Qf*conj(Kf), inverse FFT, fused scale+softmax
//   5) corrT * V (transposing multiply) -> f16
//   6) output projection WMMA GEMM + bias -> d_out (f32)
// ---------------------------------------------------------------------------

typedef _Float16 v16h __attribute__((ext_vector_type(16)));
typedef _Float16 v8h  __attribute__((ext_vector_type(8)));
typedef _Float16 v4h  __attribute__((ext_vector_type(4)));
typedef float    v8f  __attribute__((ext_vector_type(8)));

#define BATCH 4
#define SEQ   4096
#define DM    1024
#define MROWS (BATCH * SEQ)   // 16384

// ---------------- fp32 -> fp16 conversion (vectorized) ---------------------
__global__ void cvt_f32_f16_kernel(const float* __restrict__ in,
                                   _Float16* __restrict__ out, int n4) {
  int i = blockIdx.x * blockDim.x + threadIdx.x;
  int stride = gridDim.x * blockDim.x;
  for (; i < n4; i += stride) {
    float4 v = ((const float4*)in)[i];
    v4h h;
    h[0] = (_Float16)v.x; h[1] = (_Float16)v.y;
    h[2] = (_Float16)v.z; h[3] = (_Float16)v.w;
    ((v4h*)out)[i] = h;
  }
}

// ---------------- WMMA f16 GEMM:  C[M,N] = A[M,K] * B[K,N] + bias ----------
// Block: 256 threads = 8 waves; 128x128 output tile; K-step 32.
// Wave grid 4(M) x 2(N); each wave owns 32x64 = 2x4 fragments of 16x16.
// Double-buffered LDS: global loads for tile k+1 are issued before the
// WMMA work on tile k, so the s_wait_loadcnt lands after the compute.
__global__ __launch_bounds__(256)
void gemm_f16_wmma_kernel(const _Float16* __restrict__ A,
                          const _Float16* __restrict__ B,
                          const float* __restrict__ bias,
                          float* __restrict__ C,
                          int M, int N, int K) {
  constexpr int LDA = 40;  // padded row stride (halves) for 128x32 A tile
  constexpr int LDB = 40;  // padded row stride (halves) for transposed B tile
  __shared__ __align__(16) _Float16 As[2][128 * LDA];
  __shared__ __align__(16) _Float16 Bst[2][128 * LDB];   // Bst[n][k]

  const int tid  = threadIdx.x;
  const int wid  = tid >> 5;
  const int lane = tid & 31;
  const int hi   = lane >> 4;
  const int l16  = lane & 15;

  const int rowTile = blockIdx.y * 128;
  const int colTile = blockIdx.x * 128;
  const int rbase = (wid & 3) * 32;   // wave M offset
  const int cbase = (wid >> 2) * 64;  // wave N offset

  // per-thread staging coordinates (2 chunks of A, 2 chunks of B)
  const int ar0 = (tid + 0)   >> 2;          // A row, chunk 0
  const int ak0 = ((tid + 0)  & 3) * 8;      // A k-offset, chunk 0
  const int ar1 = (tid + 256) >> 2;
  const int ak1 = ((tid + 256) & 3) * 8;
  const int bk0 = (tid + 0)   >> 4;          // B k-row, chunk 0
  const int bn0 = ((tid + 0)  & 15) * 8;     // B n-offset, chunk 0
  const int bk1 = (tid + 256) >> 4;
  const int bn1 = ((tid + 256) & 15) * 8;

  const v8f zacc = {0.f, 0.f, 0.f, 0.f, 0.f, 0.f, 0.f, 0.f};
  v8f acc[2][4];
#pragma unroll
  for (int mi = 0; mi < 2; ++mi)
#pragma unroll
    for (int ni = 0; ni < 4; ++ni) acc[mi][ni] = zacc;

  v8h aReg0, aReg1, bReg0, bReg1;

  // ---- prologue: stage tile k0=0 and fill buffer 0 ----
  aReg0 = *(const v8h*)(A + (size_t)(rowTile + ar0) * K + ak0);
  aReg1 = *(const v8h*)(A + (size_t)(rowTile + ar1) * K + ak1);
  bReg0 = *(const v8h*)(B + (size_t)bk0 * N + colTile + bn0);
  bReg1 = *(const v8h*)(B + (size_t)bk1 * N + colTile + bn1);
  *(v8h*)(&As[0][ar0 * LDA + ak0]) = aReg0;
  *(v8h*)(&As[0][ar1 * LDA + ak1]) = aReg1;
#pragma unroll
  for (int j = 0; j < 8; ++j) Bst[0][(bn0 + j) * LDB + bk0] = bReg0[j];
#pragma unroll
  for (int j = 0; j < 8; ++j) Bst[0][(bn1 + j) * LDB + bk1] = bReg1[j];

  int cur = 0;
  for (int k0 = 0; k0 < K; k0 += 32) {
    __syncthreads();  // buffer `cur` is ready for everyone

    const bool more = (k0 + 32 < K);
    if (more) {  // issue next tile's global loads; no wait until after compute
      const int kn = k0 + 32;
      aReg0 = *(const v8h*)(A + (size_t)(rowTile + ar0) * K + kn + ak0);
      aReg1 = *(const v8h*)(A + (size_t)(rowTile + ar1) * K + kn + ak1);
      bReg0 = *(const v8h*)(B + (size_t)(kn + bk0) * N + colTile + bn0);
      bReg1 = *(const v8h*)(B + (size_t)(kn + bk1) * N + colTile + bn1);
      if (k0 + 64 < K) {  // pull tile-after-next toward L2 (global_prefetch_b8)
        __builtin_prefetch(A + (size_t)(rowTile + (tid >> 1)) * K + k0 + 64, 0, 0);
        __builtin_prefetch(B + (size_t)(k0 + 64 + (tid >> 3)) * N + colTile, 0, 0);
      }
    }

    // ---- build fragments per gfx1250 16-bit layouts ----
    const _Float16* as = As[cur];
    const _Float16* bs = Bst[cur];
    // A 16x32: lane row m=l16; halves at k = 8*hi..+7 and 16+8*hi..+7
    v16h af[2];
#pragma unroll
    for (int mi = 0; mi < 2; ++mi) {
      const _Float16* ap = &as[(rbase + mi * 16 + l16) * LDA + 8 * hi];
      v8h a0 = *(const v8h*)ap;
      v8h a1 = *(const v8h*)(ap + 16);
#pragma unroll
      for (int j = 0; j < 8; ++j) { af[mi][j] = a0[j]; af[mi][j + 8] = a1[j]; }
    }
    // B 32x16: lane col n=l16; halves at k = 8*hi..+7 and 16+8*hi..+7 (transposed)
    v16h bf[4];
#pragma unroll
    for (int ni = 0; ni < 4; ++ni) {
      const _Float16* bp = &bs[(cbase + ni * 16 + l16) * LDB + 8 * hi];
      v8h b0 = *(const v8h*)bp;
      v8h b1 = *(const v8h*)(bp + 16);
#pragma unroll
      for (int j = 0; j < 8; ++j) { bf[ni][j] = b0[j]; bf[ni][j + 8] = b1[j]; }
    }
#pragma unroll
    for (int mi = 0; mi < 2; ++mi)
#pragma unroll
      for (int ni = 0; ni < 4; ++ni)
        acc[mi][ni] = __builtin_amdgcn_wmma_f32_16x16x32_f16(
            false, af[mi], false, bf[ni], (short)0, acc[mi][ni], false, false);

    if (more) {  // drain staged registers into the other buffer
      const int nxt = cur ^ 1;
      *(v8h*)(&As[nxt][ar0 * LDA + ak0]) = aReg0;
      *(v8h*)(&As[nxt][ar1 * LDA + ak1]) = aReg1;
#pragma unroll
      for (int j = 0; j < 8; ++j) Bst[nxt][(bn0 + j) * LDB + bk0] = bReg0[j];
#pragma unroll
      for (int j = 0; j < 8; ++j) Bst[nxt][(bn1 + j) * LDB + bk1] = bReg1[j];
      cur = nxt;
    }
  }

  // ---- epilogue: bias add + store. D layout: n=l16, m = vgpr + 8*hi ----
#pragma unroll
  for (int ni = 0; ni < 4; ++ni) {
    int col = colTile + cbase + ni * 16 + l16;
    float bv = bias[col];
#pragma unroll
    for (int mi = 0; mi < 2; ++mi) {
#pragma unroll
      for (int r = 0; r < 8; ++r) {
        int row = rowTile + rbase + mi * 16 + hi * 8 + r;
        C[(size_t)row * N + col] = acc[mi][ni][r] + bv;
      }
    }
  }
}

// ---------------- [B,S,C] -> [B,C,S] transpose (32x32 LDS tiles) -----------
__global__ void transpose_kernel(const float* __restrict__ in,
                                 float* __restrict__ out) {
  __shared__ float tile[32][33];
  const int b = blockIdx.z;
  const int c0 = blockIdx.x * 32;
  const int s0 = blockIdx.y * 32;
  const float* ib = in + (size_t)b * SEQ * DM;
  float* ob = out + (size_t)b * SEQ * DM;
  for (int i = threadIdx.y; i < 32; i += 8)
    tile[i][threadIdx.x] = ib[(size_t)(s0 + i) * DM + c0 + threadIdx.x];
  __syncthreads();
  for (int i = threadIdx.y; i < 32; i += 8)
    ob[(size_t)(c0 + i) * SEQ + s0 + threadIdx.x] = tile[threadIdx.x][i];
}

// ---------------- per-channel FFT cross-correlation + fused softmax --------
// One workgroup per channel (b*DM + c).  Q+iK packed complex 4096-pt
// Stockham radix-2 FFT ping-ponging in LDS, Hermitian split, pointwise
// P = Qf*conj(Kf), inverse FFT, scale by N^-1.5/sqrt(Dk), softmax over t.
__global__ __launch_bounds__(256)
void fftcorr_softmax_kernel(const float* __restrict__ Qt,
                            const float* __restrict__ Kt,
                            float* __restrict__ corrT) {
  __shared__ float2 buf[2][SEQ];   // 64 KB ping-pong
  const int tid = threadIdx.x;
  const size_t ch = blockIdx.x;
  const float* q = Qt + ch * SEQ;
  const float* k = Kt + ch * SEQ;

  for (int i = tid; i < SEQ; i += 256) buf[0][i] = make_float2(q[i], k[i]);
  __syncthreads();

  int cur = 0;
  // forward FFT (sign -1), Stockham autosort, 12 stages
  for (int s = 0; s < 12; ++s) {
    const int Ns = 1 << s;
    float2* src = buf[cur];
    float2* dst = buf[cur ^ 1];
    for (int j = tid; j < SEQ / 2; j += 256) {
      float2 v0 = src[j];
      float2 v1 = src[j + SEQ / 2];
      int pos = j & (Ns - 1);
      float ang = -3.14159265358979f * (float)pos / (float)Ns;
      float sn, cs;
      __sincosf(ang, &sn, &cs);
      float2 w1 = make_float2(cs * v1.x - sn * v1.y, cs * v1.y + sn * v1.x);
      int id = ((j >> s) << (s + 1)) | pos;
      dst[id]      = make_float2(v0.x + w1.x, v0.y + w1.y);
      dst[id + Ns] = make_float2(v0.x - w1.x, v0.y - w1.y);
    }
    __syncthreads();
    cur ^= 1;
  }

  // Hermitian split of Z = FFT(q + i k); P = Qf * conj(Kf)
  {
    float2* Z = buf[cur];
    float2* P = buf[cur ^ 1];
    for (int kk = tid; kk < SEQ; kk += 256) {
      float2 zk = Z[kk];
      float2 zn = Z[(SEQ - kk) & (SEQ - 1)];
      float qr = 0.5f * (zk.x + zn.x), qi = 0.5f * (zk.y - zn.y);
      float kr = 0.5f * (zk.y + zn.y), ki = -0.5f * (zk.x - zn.x);
      P[kk] = make_float2(qr * kr + qi * ki, qi * kr - qr * ki);
    }
    __syncthreads();
    cur ^= 1;
  }

  // inverse FFT (sign +1)
  for (int s = 0; s < 12; ++s) {
    const int Ns = 1 << s;
    float2* src = buf[cur];
    float2* dst = buf[cur ^ 1];
    for (int j = tid; j < SEQ / 2; j += 256) {
      float2 v0 = src[j];
      float2 v1 = src[j + SEQ / 2];
      int pos = j & (Ns - 1);
      float ang = 3.14159265358979f * (float)pos / (float)Ns;
      float sn, cs;
      __sincosf(ang, &sn, &cs);
      float2 w1 = make_float2(cs * v1.x - sn * v1.y, cs * v1.y + sn * v1.x);
      int id = ((j >> s) << (s + 1)) | pos;
      dst[id]      = make_float2(v0.x + w1.x, v0.y + w1.y);
      dst[id + Ns] = make_float2(v0.x - w1.x, v0.y - w1.y);
    }
    __syncthreads();
    cur ^= 1;
  }

  // fused scale + softmax over the 4096 sequence positions
  // corr = Re(idft)/N^1.5 ; softmax argument = corr/sqrt(Dk=64)
  const float scale = 4.76837158203125e-7f;  // 1/(4096^1.5 * 8)
  float* red = (float*)(buf[cur ^ 1]);       // reuse pong buffer for reductions
  float vals[16];
  float lmax = -3.402823466e38f;
#pragma unroll
  for (int i = 0; i < 16; ++i) {
    vals[i] = buf[cur][tid + i * 256].x * scale;
    lmax = fmaxf(lmax, vals[i]);
  }
  red[tid] = lmax;
  __syncthreads();
  for (int off = 128; off > 0; off >>= 1) {
    if (tid < off) red[tid] = fmaxf(red[tid], red[tid + off]);
    __syncthreads();
  }
  const float gmax = red[0];
  __syncthreads();
  float lsum = 0.f;
#pragma unroll
  for (int i = 0; i < 16; ++i) {
    vals[i] = __expf(vals[i] - gmax);
    lsum += vals[i];
  }
  red[tid] = lsum;
  __syncthreads();
  for (int off = 128; off > 0; off >>= 1) {
    if (tid < off) red[tid] += red[tid + off];
    __syncthreads();
  }
  const float inv = 1.f / red[0];
#pragma unroll
  for (int i = 0; i < 16; ++i)
    corrT[ch * SEQ + tid + i * 256] = vals[i] * inv;
}

// ---------------- out_h[b,t,c] = f16( corrT[b,c,t] * V[b,t,c] ) ------------
__global__ void mult_transpose_kernel(const float* __restrict__ corrT,
                                      const float* __restrict__ V,
                                      _Float16* __restrict__ out) {
  __shared__ float tile[32][33];
  const int b = blockIdx.z;
  const int c0 = blockIdx.x * 32;
  const int s0 = blockIdx.y * 32;
  for (int i = threadIdx.y; i < 32; i += 8)  // coalesced along S
    tile[i][threadIdx.x] =
        corrT[((size_t)b * DM + c0 + i) * SEQ + s0 + threadIdx.x];
  __syncthreads();
  for (int i = threadIdx.y; i < 32; i += 8) {  // coalesced along C
    size_t idx = ((size_t)b * SEQ + s0 + i) * DM + c0 + threadIdx.x;
    out[idx] = (_Float16)(tile[threadIdx.x][i] * V[idx]);
  }
}

// ---------------------------------------------------------------------------
extern "C" void kernel_launch(void* const* d_in, const int* in_sizes, int n_in,
                              void* d_out, int out_size, void* d_ws,
                              size_t ws_size, hipStream_t stream) {
  (void)in_sizes; (void)n_in; (void)out_size; (void)ws_size;
  const float* x  = (const float*)d_in[0];
  const float* Wq = (const float*)d_in[1];
  const float* bq = (const float*)d_in[2];
  const float* Wk = (const float*)d_in[3];
  const float* bk = (const float*)d_in[4];
  const float* Wv = (const float*)d_in[5];
  const float* bv = (const float*)d_in[6];
  const float* Wo = (const float*)d_in[7];
  const float* bo = (const float*)d_in[8];
  float* out = (float*)d_out;

  // workspace layout (~296 MiB)
  char* p = (char*)d_ws;
  _Float16* xh  = (_Float16*)p; p += (size_t)MROWS * DM * 2;  // 32 MB
  _Float16* wqh = (_Float16*)p; p += (size_t)DM * DM * 2;
  _Float16* wkh = (_Float16*)p; p += (size_t)DM * DM * 2;
  _Float16* wvh = (_Float16*)p; p += (size_t)DM * DM * 2;
  _Float16* woh = (_Float16*)p; p += (size_t)DM * DM * 2;
  float* Qs = (float*)p; p += (size_t)MROWS * DM * 4;  // reused for K
  float* Vs = (float*)p; p += (size_t)MROWS * DM * 4;
  float* Qt = (float*)p; p += (size_t)MROWS * DM * 4;  // corrT in place
  float* Kt = (float*)p; p += (size_t)MROWS * DM * 4;
  _Float16* mh = xh;  // reuse x's f16 copy for corr*V output

  const dim3 gGemm(DM / 128, MROWS / 128);   // (8,128)
  const dim3 gTile(DM / 32, SEQ / 32, BATCH);
  const dim3 bTile(32, 8);

  // 1) f16 conversions
  cvt_f32_f16_kernel<<<4096, 256, 0, stream>>>(x, xh, MROWS * DM / 4);
  cvt_f32_f16_kernel<<<1024, 256, 0, stream>>>(Wq, wqh, DM * DM / 4);
  cvt_f32_f16_kernel<<<1024, 256, 0, stream>>>(Wk, wkh, DM * DM / 4);
  cvt_f32_f16_kernel<<<1024, 256, 0, stream>>>(Wv, wvh, DM * DM / 4);
  cvt_f32_f16_kernel<<<1024, 256, 0, stream>>>(Wo, woh, DM * DM / 4);

  // 2) projections + transposes
  gemm_f16_wmma_kernel<<<gGemm, 256, 0, stream>>>(xh, wqh, bq, Qs, MROWS, DM, DM);
  transpose_kernel<<<gTile, bTile, 0, stream>>>(Qs, Qt);
  gemm_f16_wmma_kernel<<<gGemm, 256, 0, stream>>>(xh, wkh, bk, Qs, MROWS, DM, DM);
  transpose_kernel<<<gTile, bTile, 0, stream>>>(Qs, Kt);
  gemm_f16_wmma_kernel<<<gGemm, 256, 0, stream>>>(xh, wvh, bv, Vs, MROWS, DM, DM);

  // 3) FFT correlation + fused softmax (corrT overwrites Qt)
  fftcorr_softmax_kernel<<<BATCH * DM, 256, 0, stream>>>(Qt, Kt, Qt);

  // 4) corr * V -> f16
  mult_transpose_kernel<<<gTile, bTile, 0, stream>>>(Qt, Vs, mh);

  // 5) output projection -> d_out
  gemm_f16_wmma_kernel<<<gGemm, 256, 0, stream>>>(mh, woh, bo, out, MROWS, DM, DM);
}
